// TimeAwareAttention_49744311222623
// MI455X (gfx1250) — matrix operand order repaired and checked by reading, not compiled, not run on hardware
//
#include <hip/hip_runtime.h>
#include <hip/hip_bf16.h>

// ---------------------------------------------------------------------------
// TimeAwareAttention for MI455X (gfx1250), wave32 + WMMA bf16.
// B=4, S=2048, D=512, H=8, DH=64, MAX_LEN=4096
// ---------------------------------------------------------------------------

typedef __bf16 bf16_t;
typedef __attribute__((ext_vector_type(16))) bf16_t v16bf;
typedef __attribute__((ext_vector_type(8)))  bf16_t v8bf;
typedef __attribute__((ext_vector_type(8)))  float  v8f;

#define Bn   4
#define Sn   2048
#define Dn   512
#define Hn   8
#define DHn  64
#define MAXL 4096

static __device__ __forceinline__ bf16_t f2bf(float f) {
  unsigned u = __float_as_uint(f);
  unsigned r = u + 0x7FFFu + ((u >> 16) & 1u);
  unsigned short h = (unsigned short)(r >> 16);
  return __builtin_bit_cast(bf16_t, h);
}

static __device__ __forceinline__ v8f wmma_bf16(v16bf a, v16bf b, v8f c) {
  // D = A(16x32 bf16) * B(32x16 bf16) + C(16x16 f32)
  return __builtin_amdgcn_wmma_f32_16x16x32_bf16(
      false, a, false, b, (short)0, c, false, false);
}

// ---------------------------------------------------------------------------
// LUT:  lut[d + 4095] = sum_k relu(d*w1[k] + b1[k]) * w2[k] + b2
// ---------------------------------------------------------------------------
__global__ void taa_lut_kernel(const float* __restrict__ w1,
                               const float* __restrict__ b1,
                               const float* __restrict__ w2,
                               const float* __restrict__ b2,
                               float* __restrict__ lut) {
  int idx = blockIdx.x * blockDim.x + threadIdx.x;
  if (idx >= 2 * MAXL - 1) return;
  float d = (float)(idx - (MAXL - 1));
  float acc = 0.f;
#pragma unroll 8
  for (int k = 0; k < DHn; ++k) {
    float hv = fmaxf(fmaf(d, w1[k], b1[k]), 0.f);
    acc = fmaf(hv, w2[k], acc);
  }
  lut[idx] = acc + b2[0];
}

// ---------------------------------------------------------------------------
// fp32 -> bf16 elementwise
// ---------------------------------------------------------------------------
__global__ void taa_cvt_kernel(const float* __restrict__ src,
                               bf16_t* __restrict__ dst, int n) {
  int i = blockIdx.x * blockDim.x + threadIdx.x;
  if (i < n) dst[i] = f2bf(src[i]);
}

// W[k][n] (512x512 fp32) -> Wt[n][k] bf16 (transpose+convert)
__global__ void taa_trcvt_kernel(const float* __restrict__ W,
                                 bf16_t* __restrict__ Wt) {
  int t = blockIdx.x * blockDim.x + threadIdx.x; // 262144
  int n = t >> 9, k = t & 511;
  Wt[t] = f2bf(W[(size_t)k * Dn + n]);
}

// ---------------------------------------------------------------------------
// QKV GEMM:  [8192,512]bf16 x Wt[1536,512]bf16 -> q,k [B,H,S,DH], vT [B,H,DH,S]
// block = 128 thr (4 waves), wave: 16 rows x 64 cols, grid (128, 24)
// ---------------------------------------------------------------------------
__global__ void taa_qkv_gemm_kernel(const bf16_t* __restrict__ Qb,
                                    const bf16_t* __restrict__ Wt,
                                    const float* __restrict__ bq,
                                    const float* __restrict__ bk,
                                    const float* __restrict__ bv,
                                    bf16_t* __restrict__ qh,
                                    bf16_t* __restrict__ kh,
                                    bf16_t* __restrict__ vt) {
  const int wave = threadIdx.x >> 5, lane = threadIdx.x & 31;
  const int il = lane & 15, lh = lane >> 4;
  const int m0 = blockIdx.x * 64 + wave * 16;
  const int n0 = blockIdx.y * 64;                 // global col in [0,1536)
  const int which = n0 >> 9;                      // 0=q 1=k 2=v
  const int nin = n0 & 511;
  const int hh = nin >> 6;

  v8f acc[4] = {};
  const bf16_t* arow = Qb + (size_t)(m0 + il) * Dn;
#pragma unroll 4
  for (int k0 = 0; k0 < Dn; k0 += 32) {
    const bf16_t* ap = arow + k0 + 16 * lh;
    v8bf lo = *(const v8bf*)ap;
    v8bf hi = *(const v8bf*)(ap + 8);
    v16bf a;
#pragma unroll
    for (int e = 0; e < 8; ++e) { a[e] = lo[e]; a[e + 8] = hi[e]; }
#pragma unroll
    for (int t = 0; t < 4; ++t) {
      const bf16_t* bp = Wt + (size_t)(n0 + t * 16 + il) * Dn + k0 + 16 * lh;
      v16bf bfr = *(const v16bf*)bp;
      acc[t] = wmma_bf16(a, bfr, acc[t]);
    }
  }

  const float* bias = (which == 0) ? bq : (which == 1) ? bk : bv;
#pragma unroll
  for (int t = 0; t < 4; ++t) {
    int n = nin + t * 16 + il;
    int dh = n & 63;
    float bs = bias[n];
#pragma unroll
    for (int r = 0; r < 8; ++r) {
      int m = m0 + r + 8 * lh;
      int bb = m >> 11, ss = m & (Sn - 1);
      size_t bh = (size_t)(bb * Hn + hh);
      float val = acc[t][r] + bs;
      if (which == 0)      qh[(bh * Sn + ss) * DHn + dh] = f2bf(val * 0.125f);
      else if (which == 1) kh[(bh * Sn + ss) * DHn + dh] = f2bf(val);
      else                 vt[(bh * DHn + dh) * Sn + ss] = f2bf(val);
    }
  }
}

// ---------------------------------------------------------------------------
// Flash attention.  grid (S/64, H, B), block 128 (4 waves, 16 q-rows each).
// Computes S^T tiles = K * Q^T so softmax rows live per-lane and the fp32
// C-fragment converts in-lane into the bf16 A-fragment for P*V.
// ---------------------------------------------------------------------------
__global__ void taa_attn_kernel(const bf16_t* __restrict__ qh,
                                const bf16_t* __restrict__ kh,
                                const bf16_t* __restrict__ vt,
                                const int* __restrict__ tidx,
                                const float* __restrict__ trel,
                                const float* __restrict__ lut,
                                bf16_t* __restrict__ ctxws) {
  const int wave = threadIdx.x >> 5, lane = threadIdx.x & 31;
  const int il = lane & 15, lh = lane >> 4;
  const int b = blockIdx.z, h = blockIdx.y;
  const int i0 = blockIdx.x * 64 + wave * 16;
  const size_t bh = (size_t)(b * Hn + h);

  // q B-fragments (K = dh): lane holds col i = il, elements k = dh0+16*lh+e
  const bf16_t* qrow = qh + (bh * Sn + i0 + il) * DHn;
  v16bf qf0 = *(const v16bf*)(qrow + 16 * lh);
  v16bf qf1 = *(const v16bf*)(qrow + 32 + 16 * lh);

  const int ti = tidx[b * Sn + i0 + il];
  const bf16_t* kbase = kh + bh * (size_t)Sn * DHn;
  const bf16_t* vbase = vt + bh * (size_t)DHn * Sn;
  const int* tjb = tidx + b * Sn;
  const float* trow = trel + (size_t)(i0 + il) * MAXL;

  float m_run = -1e30f, l_run = 0.f;
  v8f cx[4] = {};

  for (int j0 = 0; j0 < Sn; j0 += 32) {
    // ---- prefetch next j-block (k rows, v^T segments, trel segment) ----
    if (j0 + 32 < Sn) {
      __builtin_prefetch(kbase + (size_t)(j0 + 32 + lane) * DHn, 0, 1);
      __builtin_prefetch(kbase + (size_t)(j0 + 32 + lane) * DHn + 32, 0, 1);
#pragma unroll
      for (int f = 0; f < 4; ++f)
        __builtin_prefetch(vbase + (size_t)(f * 16 + il) * Sn + j0 + 32 +
                               16 * lh, 0, 1);
      __builtin_prefetch(trow + j0 + 32 + 8 * lh, 0, 1);
    }

    // ---- S^T tiles: rows j, cols i ----
    v8f s0, s1;
#pragma unroll
    for (int jt = 0; jt < 2; ++jt) {
      const bf16_t* kp = kbase + (size_t)(j0 + jt * 16 + il) * DHn + 16 * lh;
      v8bf x0 = *(const v8bf*)(kp);
      v8bf x1 = *(const v8bf*)(kp + 8);
      v8bf x2 = *(const v8bf*)(kp + 32);
      v8bf x3 = *(const v8bf*)(kp + 40);
      v16bf a0, a1;
#pragma unroll
      for (int e = 0; e < 8; ++e) {
        a0[e] = x0[e]; a0[e + 8] = x1[e];
        a1[e] = x2[e]; a1[e + 8] = x3[e];
      }
      v8f c = {};
      c = wmma_bf16(a0, qf0, c);
      c = wmma_bf16(a1, qf1, c);
      if (jt == 0) s0 = c; else s1 = c;
    }

    // ---- add time_relation + LUT(time diff) bias; tile maxes ----
    float mx = -1e30f;
#pragma unroll
    for (int jt = 0; jt < 2; ++jt) {
      int jb = j0 + jt * 16 + 8 * lh;
      int4 ta = *(const int4*)(tjb + jb);
      int4 tb = *(const int4*)(tjb + jb + 4);
      int tj[8] = {ta.x, ta.y, ta.z, ta.w, tb.x, tb.y, tb.z, tb.w};
#pragma unroll
      for (int r = 0; r < 8; ++r) {
        float bias = trow[jb + r] + lut[ti - tj[r] + (MAXL - 1)];
        if (jt == 0) { s0[r] += bias; mx = fmaxf(mx, s0[r]); }
        else         { s1[r] += bias; mx = fmaxf(mx, s1[r]); }
      }
    }
    mx = fmaxf(mx, __shfl_xor(mx, 16, 32));
    float m_new = fmaxf(m_run, mx);
    float alpha = __expf(m_run - m_new);

    // ---- p = exp(s - m), packed directly into bf16 A-fragment ----
    v16bf p;
    float rs = 0.f;
#pragma unroll
    for (int r = 0; r < 8; ++r) {
      float e0 = __expf(s0[r] - m_new);
      float e1 = __expf(s1[r] - m_new);
      rs += e0 + e1;
      p[r]     = f2bf(e0);
      p[r + 8] = f2bf(e1);
    }
    rs += __shfl_xor(rs, 16, 32);
    l_run = l_run * alpha + rs;
    m_run = m_new;

    // ---- rescale ctx rows (row index = r + 8*lh) ----
    float ar[8];
#pragma unroll
    for (int r = 0; r < 8; ++r) ar[r] = __shfl(alpha, r + 8 * lh, 32);
#pragma unroll
    for (int f = 0; f < 4; ++f)
#pragma unroll
      for (int r = 0; r < 8; ++r) cx[f][r] *= ar[r];

    // ---- ctx += P * V  (V^T stored [dh][s] => contiguous B-frag) ----
#pragma unroll
    for (int f = 0; f < 4; ++f) {
      const bf16_t* vp = vbase + (size_t)(f * 16 + il) * Sn + j0 + 8 * lh;
      v8bf vlo = *(const v8bf*)vp;
      v8bf vhi = *(const v8bf*)(vp + 16);
      v16bf bv;
#pragma unroll
      for (int e = 0; e < 8; ++e) { bv[e] = vlo[e]; bv[e + 8] = vhi[e]; }
      cx[f] = wmma_bf16(p, bv, cx[f]);
    }
  }

  // ---- finalize: divide by row sums, store ctx [B,S,H*DH] bf16 ----
  float linv = 1.f / l_run;
  float lr[8];
#pragma unroll
  for (int r = 0; r < 8; ++r) lr[r] = __shfl(linv, r + 8 * lh, 32);
#pragma unroll
  for (int f = 0; f < 4; ++f)
#pragma unroll
    for (int r = 0; r < 8; ++r) {
      int irow = i0 + r + 8 * lh;
      ctxws[((size_t)(b * Sn + irow)) * Dn + h * DHn + f * 16 + il] =
          f2bf(cx[f][r] * lr[r]);
    }
}

// ---------------------------------------------------------------------------
// Output projection + bias + residual -> xws fp32.  grid (128, 8), block 128.
// ---------------------------------------------------------------------------
__global__ void taa_out_gemm_kernel(const bf16_t* __restrict__ ctxws,
                                    const bf16_t* __restrict__ fcWt,
                                    const float* __restrict__ fcb,
                                    const float* __restrict__ Qin,
                                    float* __restrict__ xws) {
  const int wave = threadIdx.x >> 5, lane = threadIdx.x & 31;
  const int il = lane & 15, lh = lane >> 4;
  const int m0 = blockIdx.x * 64 + wave * 16;
  const int n0 = blockIdx.y * 64;

  v8f acc[4] = {};
  const bf16_t* arow = ctxws + (size_t)(m0 + il) * Dn;
#pragma unroll 4
  for (int k0 = 0; k0 < Dn; k0 += 32) {
    const bf16_t* ap = arow + k0 + 16 * lh;
    v8bf lo = *(const v8bf*)ap;
    v8bf hi = *(const v8bf*)(ap + 8);
    v16bf a;
#pragma unroll
    for (int e = 0; e < 8; ++e) { a[e] = lo[e]; a[e + 8] = hi[e]; }
#pragma unroll
    for (int t = 0; t < 4; ++t) {
      const bf16_t* bp = fcWt + (size_t)(n0 + t * 16 + il) * Dn + k0 + 16 * lh;
      v16bf bfr = *(const v16bf*)bp;
      acc[t] = wmma_bf16(a, bfr, acc[t]);
    }
  }
#pragma unroll
  for (int t = 0; t < 4; ++t) {
    int n = n0 + t * 16 + il;
    float bs = fcb[n];
#pragma unroll
    for (int r = 0; r < 8; ++r) {
      size_t m = (size_t)(m0 + r + 8 * lh);
      xws[m * Dn + n] = acc[t][r] + bs + Qin[m * Dn + n];
    }
  }
}

// ---------------------------------------------------------------------------
// LayerNorm: wave per row (512 elems = 16/lane). grid 1024, block 256.
// ---------------------------------------------------------------------------
__global__ void taa_ln_kernel(const float* __restrict__ xws,
                              const float* __restrict__ g,
                              const float* __restrict__ bt,
                              float* __restrict__ out) {
  const int wave = threadIdx.x >> 5, lane = threadIdx.x & 31;
  const size_t row = (size_t)blockIdx.x * 8 + wave;
  const float* rp = xws + row * Dn;
  float x[16];
  float s = 0.f;
#pragma unroll
  for (int e = 0; e < 16; ++e) { x[e] = rp[e * 32 + lane]; s += x[e]; }
#pragma unroll
  for (int m = 16; m >= 1; m >>= 1) s += __shfl_xor(s, m, 32);
  float mu = s * (1.f / 512.f);
  float vs = 0.f;
#pragma unroll
  for (int e = 0; e < 16; ++e) { float d = x[e] - mu; vs += d * d; }
#pragma unroll
  for (int m = 16; m >= 1; m >>= 1) vs += __shfl_xor(vs, m, 32);
  float rstd = rsqrtf(vs * (1.f / 512.f) + 1e-5f);
  float* op = out + row * Dn;
#pragma unroll
  for (int e = 0; e < 16; ++e) {
    int c = e * 32 + lane;
    op[c] = (x[e] - mu) * rstd * g[c] + bt[c];
  }
}

// ---------------------------------------------------------------------------
// Launch
// ---------------------------------------------------------------------------
extern "C" void kernel_launch(void* const* d_in, const int* in_sizes, int n_in,
                              void* d_out, int out_size, void* d_ws,
                              size_t ws_size, hipStream_t stream) {
  (void)in_sizes; (void)n_in; (void)out_size; (void)ws_size;
  const float* Q    = (const float*)d_in[0];
  const int*   tix  = (const int*)d_in[1];
  const float* Wq   = (const float*)d_in[2];
  const float* bq   = (const float*)d_in[3];
  const float* Wk   = (const float*)d_in[4];
  const float* bk   = (const float*)d_in[5];
  const float* Wv   = (const float*)d_in[6];
  const float* bv   = (const float*)d_in[7];
  const float* trel = (const float*)d_in[8];
  const float* tpw1 = (const float*)d_in[9];
  const float* tpb1 = (const float*)d_in[10];
  const float* tpw2 = (const float*)d_in[11];
  const float* tpb2 = (const float*)d_in[12];
  const float* fcw  = (const float*)d_in[13];
  const float* fcb  = (const float*)d_in[14];
  const float* lng  = (const float*)d_in[15];
  const float* lnb  = (const float*)d_in[16];
  float* out = (float*)d_out;

  char* ws = (char*)d_ws;
  const size_t MSZ = (size_t)Bn * Sn * Dn;           // 4194304 elems
  size_t off = 0;
  float*  lut   = (float*)(ws + off);  off += 8192 * sizeof(float);
  bf16_t* Qb    = (bf16_t*)(ws + off); off += MSZ * 2;
  bf16_t* Wt    = (bf16_t*)(ws + off); off += (size_t)3 * Dn * Dn * 2;
  bf16_t* fcWt  = (bf16_t*)(ws + off); off += (size_t)Dn * Dn * 2;
  bf16_t* qhd   = (bf16_t*)(ws + off); off += MSZ * 2;
  bf16_t* khd   = (bf16_t*)(ws + off); off += MSZ * 2;
  bf16_t* vtd   = (bf16_t*)(ws + off); off += MSZ * 2;
  bf16_t* ctxws = (bf16_t*)(ws + off); off += MSZ * 2;
  float*  xws   = (float*)(ws + off);  off += MSZ * sizeof(float);

  taa_lut_kernel<<<32, 256, 0, stream>>>(tpw1, tpb1, tpw2, tpb2, lut);
  taa_cvt_kernel<<<(int)(MSZ / 256), 256, 0, stream>>>(Q, Qb, (int)MSZ);
  taa_trcvt_kernel<<<1024, 256, 0, stream>>>(Wq, Wt);
  taa_trcvt_kernel<<<1024, 256, 0, stream>>>(Wk, Wt + (size_t)Dn * Dn);
  taa_trcvt_kernel<<<1024, 256, 0, stream>>>(Wv, Wt + (size_t)2 * Dn * Dn);
  taa_trcvt_kernel<<<1024, 256, 0, stream>>>(fcw, fcWt);

  taa_qkv_gemm_kernel<<<dim3(128, 24), 128, 0, stream>>>(Qb, Wt, bq, bk, bv,
                                                         qhd, khd, vtd);
  taa_attn_kernel<<<dim3(Sn / 64, Hn, Bn), 128, 0, stream>>>(
      qhd, khd, vtd, tix, trel, lut, ctxws);
  taa_out_gemm_kernel<<<dim3(128, 8), 128, 0, stream>>>(ctxws, fcWt, fcb, Q,
                                                        xws);
  taa_ln_kernel<<<1024, 256, 0, stream>>>(xws, lng, lnb, out);
}
